// DEQCfCSequenceProcessor_36232344109584
// MI455X (gfx1250) — matrix operand order repaired and checked by reading, not compiled, not run on hardware
//
#include <hip/hip_runtime.h>
#include <hip/hip_bf16.h>
#include <stdint.h>

#define BATCH  64
#define SEQ    1024
#define DMODEL 1024
#define HID    1024
#define NGATE  4
#define OUT_MAIN (BATCH*SEQ*HID)   // 67108864 floats, then final_state

typedef __attribute__((ext_vector_type(16))) __bf16 v16bf;
typedef __attribute__((ext_vector_type(8)))  float  v8f;
typedef __attribute__((vector_size(4 * sizeof(int)))) int v4i_t;
typedef __attribute__((address_space(1))) v4i_t gv4i_t;   // global
typedef __attribute__((address_space(3))) v4i_t lv4i_t;   // LDS

#if defined(__has_builtin)
#  if __has_builtin(__builtin_amdgcn_global_load_async_to_lds_b128) && \
      __has_builtin(__builtin_amdgcn_s_wait_asynccnt)
#    define HAVE_ASYNC 1
#  endif
#endif
#ifndef HAVE_ASYNC
#  define HAVE_ASYNC 0
#endif

static __device__ __forceinline__ unsigned short f2bf(float f) {
  unsigned u = __float_as_uint(f);
  u += 0x7fffu + ((u >> 16) & 1u);      // round-to-nearest-even
  return (unsigned short)(u >> 16);
}
static __device__ __forceinline__ unsigned pk2bf(float a, float b) {
  return (unsigned)f2bf(a) | ((unsigned)f2bf(b) << 16);
}

// ---------------------------------------------------------------------------
// Kernel 0: zero barrier counter, convert init_state -> bf16 H0.
// ---------------------------------------------------------------------------
__global__ __launch_bounds__(256) void init_state_kernel(
    const float* __restrict__ h0, unsigned short* __restrict__ H0,
    unsigned* __restrict__ cnt) {
  unsigned i = blockIdx.x * 256u + threadIdx.x;
  if (i == 0) *cnt = 0u;
  if (i < BATCH * HID) H0[i] = f2bf(h0[i]);
}

// ---------------------------------------------------------------------------
// Kernel 1: pack four fp32 weights (2048x1024) into bf16 WMMA-B fragment
// order: lane L holds n = nt*16+(L&15); dword v holds K = kt*32 +
// 16*(v>>2) + (L<16?0:8) + 2*(v&3)..+1.  Wp[(((g*64+nt)*64+kt)*32+lane)*8+v] -> 2x b128/lane.
// ---------------------------------------------------------------------------
__global__ __launch_bounds__(256) void pack_weights(
    const float* __restrict__ W1, const float* __restrict__ W2,
    const float* __restrict__ Wta, const float* __restrict__ Wtb,
    unsigned* __restrict__ Wp) {
  unsigned gtid = blockIdx.x * 256u + threadIdx.x;   // 0 .. 4194303
  unsigned v    = gtid & 7u;
  unsigned lane = (gtid >> 3) & 31u;
  unsigned tile = gtid >> 8;
  unsigned kt = tile & 63u, nt = (tile >> 6) & 63u, g = tile >> 12;
  const float* W = (g == 0) ? W1 : (g == 1) ? W2 : (g == 2) ? Wta : Wtb;
  unsigned k = kt * 32u + ((v & 4u) << 2) + ((lane >> 4) << 3) + ((v & 3u) << 1);
  unsigned n = nt * 16u + (lane & 15u);
  Wp[gtid] = pk2bf(W[(size_t)k * HID + n], W[(size_t)(k + 1) * HID + n]);
}

// ---------------------------------------------------------------------------
// Kernel 1b: bulk fp32 -> bf16 conversion of ssm_output (bandwidth job,
// off the sequential critical path).  8 elements / thread, b128 in/out.
// ---------------------------------------------------------------------------
__global__ __launch_bounds__(256) void convert_x(
    const float* __restrict__ X, unsigned short* __restrict__ Xb) {
  size_t i = ((size_t)blockIdx.x * 256u + threadIdx.x) * 8u;
  float4 a = *(const float4*)(X + i);
  float4 b = *(const float4*)(X + i + 4);
  uint4 o;
  o.x = pk2bf(a.x, a.y); o.y = pk2bf(a.z, a.w);
  o.z = pk2bf(b.x, b.y); o.w = pk2bf(b.z, b.w);
  *(uint4*)(Xb + i) = o;
}

// ---------------------------------------------------------------------------
// Shared epilogue + barrier helpers for the scan kernels.
// ---------------------------------------------------------------------------
static __device__ __forceinline__ void cfc_epilogue(
    const v8f acc[NGATE], int wave, int lane, int j, int t,
    float bb1, float bb2, float bbta, float bbtb, const float* dts,
    float* __restrict__ out, unsigned short* __restrict__ Hn) {
#pragma unroll
  for (int v = 0; v < 8; ++v) {
    int brow = wave * 16 + v + ((lane >> 4) << 3);   // batch index (C: M=v+8*hi)
    float f1 = tanhf(acc[0][v] + bb1);
    float f2 = tanhf(acc[1][v] + bb2);
    float ta = acc[2][v] + bbta;
    float tb = acc[3][v] + bbtb;
    float s  = 1.f / (1.f + __expf(-(ta * dts[brow] + tb)));
    float hn = f1 * (1.f - s) + s * f2;
    out[((size_t)brow * SEQ + t) * HID + j] = hn;
    if (t == SEQ - 1)
      out[(size_t)OUT_MAIN + (size_t)brow * HID + j] = hn;
    Hn[(size_t)brow * HID + j] = f2bf(hn);
  }
}

static __device__ __forceinline__ void grid_step_barrier(
    unsigned* cnt, int tid, int t) {
  __threadfence();
  __syncthreads();
  if (tid == 0) {
    atomicAdd(cnt, 1u);
    unsigned target = 64u * (unsigned)(t + 1);
    while (__atomic_load_n(cnt, __ATOMIC_ACQUIRE) < target)
      __builtin_amdgcn_s_sleep(2);
  }
  __syncthreads();
  __threadfence();
}

// ---------------------------------------------------------------------------
// Kernel 2a: persistent CfC scan, pre-converted bf16 X.  64 WGs x 128 thr
// (4 waves).  Double-buffered LDS chunks streamed with async-to-LDS DMA:
// issue chunk c+1, s_wait_asynccnt(16) (in-order) => chunk c landed, overlap
// the copy of c+1 with the 32 wmma of chunk c.
// ---------------------------------------------------------------------------
__global__ __launch_bounds__(128) void cfc_scan_pre(
    const unsigned short* __restrict__ Xb, const float* __restrict__ dt,
    const float* __restrict__ b1, const float* __restrict__ b2,
    const float* __restrict__ bta, const float* __restrict__ btb,
    const unsigned* __restrict__ Wp,
    unsigned short* __restrict__ H0, unsigned short* __restrict__ H1,
    float* __restrict__ out, unsigned* __restrict__ cnt) {
  __shared__ __align__(16) unsigned short zs[2][64][264];  // double buffer
  __shared__ float dts[BATCH];

  const int tid  = threadIdx.x;
  const int lane = tid & 31;
  const int wave = tid >> 5;
  const int wg   = blockIdx.x;
  const int jn   = lane & 15;
  const int j    = wg * 16 + jn;
  const float bb1 = b1[j], bb2 = b2[j], bbta = bta[j], bbtb = btb[j];
  if (tid < BATCH) dts[tid] = dt[tid];
  const int r  = tid >> 1;
  const int hp = tid & 1;
  const int arow = wave * 16 + jn;
  const int kgrp = (lane >> 4) << 3;

  for (int t = 0; t < SEQ; ++t) {
    const unsigned short* __restrict__ Hc = (t & 1) ? H1 : H0;
    unsigned short* __restrict__ Hn = (t & 1) ? H0 : H1;

    auto issue = [&](int c, int buf) {
      const unsigned short* src = (c < 4)
          ? Xb + ((size_t)r * SEQ + t) * DMODEL + c * 256 + hp * 128
          : Hc + (size_t)r * HID + (c - 4) * 256 + hp * 128;
      unsigned short* dst = &zs[buf][r][hp * 128];
#if HAVE_ASYNC
#pragma unroll
      for (int i = 0; i < 16; ++i)
        __builtin_amdgcn_global_load_async_to_lds_b128(
            (gv4i_t*)(src + i * 8), (lv4i_t*)(dst + i * 8), 0, 0);
#else
#pragma unroll
      for (int i = 0; i < 16; ++i)
        ((uint4*)dst)[i] = ((const uint4*)src)[i];
#endif
    };

    v8f acc[NGATE];
#pragma unroll
    for (int g = 0; g < NGATE; ++g)
      acc[g] = (v8f){0.f, 0.f, 0.f, 0.f, 0.f, 0.f, 0.f, 0.f};

    issue(0, 0);
    for (int c = 0; c < 8; ++c) {
      if (c < 7) issue(c + 1, (c + 1) & 1);
#if HAVE_ASYNC
      if (c < 7) __builtin_amdgcn_s_wait_asynccnt(16);
      else       __builtin_amdgcn_s_wait_asynccnt(0);
#endif
      __syncthreads();                 // chunk c visible to all waves
      const int buf = c & 1;
#pragma unroll
      for (int kt = 0; kt < 8; ++kt) {
        union { unsigned u[8]; v16bf v; } af;
#pragma unroll
        for (int v = 0; v < 8; ++v) {
          int k0 = kt * 32 + ((v & 4) << 2) + kgrp + ((v & 3) << 1);
          af.u[v] = *(const unsigned*)&zs[buf][arow][k0];
        }
        const int ktg = c * 8 + kt;
#pragma unroll
        for (int g = 0; g < NGATE; ++g) {
          union { uint4 q[2]; v16bf v; } bfr;
          const uint4* bp = (const uint4*)(
              Wp + ((((size_t)g * 64 + wg) * 64 + ktg) * 32 + lane) * 8);
          bfr.q[0] = bp[0];
          bfr.q[1] = bp[1];
          acc[g] = __builtin_amdgcn_wmma_f32_16x16x32_bf16(
              false, af.v, false, bfr.v, (short)0, acc[g], false, false);
        }
      }
      __syncthreads();                 // done reading buf before re-fill
    }

    cfc_epilogue(acc, wave, lane, j, t, bb1, bb2, bbta, bbtb, dts, out, Hn);
    grid_step_barrier(cnt, tid, t);
  }
}

// ---------------------------------------------------------------------------
// Kernel 2b: fallback scan (small workspace): convert x_t in-scan,
// single-buffered staging.
// ---------------------------------------------------------------------------
__global__ __launch_bounds__(128) void cfc_scan(
    const float* __restrict__ X, const float* __restrict__ dt,
    const float* __restrict__ b1, const float* __restrict__ b2,
    const float* __restrict__ bta, const float* __restrict__ btb,
    const unsigned* __restrict__ Wp,
    unsigned short* __restrict__ H0, unsigned short* __restrict__ H1,
    float* __restrict__ out, unsigned* __restrict__ cnt) {
  __shared__ __align__(16) unsigned short zs[64][264];
  __shared__ float dts[BATCH];

  const int tid  = threadIdx.x;
  const int lane = tid & 31;
  const int wave = tid >> 5;
  const int wg   = blockIdx.x;
  const int jn   = lane & 15;
  const int j    = wg * 16 + jn;
  const float bb1 = b1[j], bb2 = b2[j], bbta = bta[j], bbtb = btb[j];
  if (tid < BATCH) dts[tid] = dt[tid];
  const int r  = tid >> 1;
  const int hp = tid & 1;
  const int arow = wave * 16 + jn;
  const int kgrp = (lane >> 4) << 3;

  for (int t = 0; t < SEQ; ++t) {
    const unsigned short* __restrict__ Hc = (t & 1) ? H1 : H0;
    unsigned short* __restrict__ Hn = (t & 1) ? H0 : H1;

    v8f acc[NGATE];
#pragma unroll
    for (int g = 0; g < NGATE; ++g)
      acc[g] = (v8f){0.f, 0.f, 0.f, 0.f, 0.f, 0.f, 0.f, 0.f};

    for (int c = 0; c < 8; ++c) {
      __syncthreads();
      if (c < 4) {                     // x_t: fp32 -> bf16 convert
        const float4* src =
            (const float4*)(X + ((size_t)r * SEQ + t) * DMODEL + c * 256 + hp * 128);
        unsigned* dstw = (unsigned*)&zs[r][hp * 128];
#pragma unroll
        for (int i = 0; i < 32; ++i) {
          float4 f = src[i];
          dstw[2 * i + 0] = pk2bf(f.x, f.y);
          dstw[2 * i + 1] = pk2bf(f.z, f.w);
        }
      } else {                         // h part: bf16 async copy
        const unsigned short* hsrc = Hc + (size_t)r * HID + (c - 4) * 256 + hp * 128;
        unsigned short* dst = &zs[r][hp * 128];
#if HAVE_ASYNC
#pragma unroll
        for (int i = 0; i < 16; ++i)
          __builtin_amdgcn_global_load_async_to_lds_b128(
              (gv4i_t*)(hsrc + i * 8), (lv4i_t*)(dst + i * 8), 0, 0);
        __builtin_amdgcn_s_wait_asynccnt(0);
#else
#pragma unroll
        for (int i = 0; i < 16; ++i)
          ((uint4*)dst)[i] = ((const uint4*)hsrc)[i];
#endif
      }
      __syncthreads();

#pragma unroll
      for (int kt = 0; kt < 8; ++kt) {
        union { unsigned u[8]; v16bf v; } af;
#pragma unroll
        for (int v = 0; v < 8; ++v) {
          int k0 = kt * 32 + ((v & 4) << 2) + kgrp + ((v & 3) << 1);
          af.u[v] = *(const unsigned*)&zs[arow][k0];
        }
        const int ktg = c * 8 + kt;
#pragma unroll
        for (int g = 0; g < NGATE; ++g) {
          union { uint4 q[2]; v16bf v; } bfr;
          const uint4* bp = (const uint4*)(
              Wp + ((((size_t)g * 64 + wg) * 64 + ktg) * 32 + lane) * 8);
          bfr.q[0] = bp[0];
          bfr.q[1] = bp[1];
          acc[g] = __builtin_amdgcn_wmma_f32_16x16x32_bf16(
              false, af.v, false, bfr.v, (short)0, acc[g], false, false);
        }
      }
    }

    cfc_epilogue(acc, wave, lane, j, t, bb1, bb2, bbta, bbtb, dts, out, Hn);
    grid_step_barrier(cnt, tid, t);
  }
}

// ---------------------------------------------------------------------------
extern "C" void kernel_launch(void* const* d_in, const int* in_sizes, int n_in,
                              void* d_out, int out_size, void* d_ws, size_t ws_size,
                              hipStream_t stream) {
  (void)in_sizes; (void)n_in; (void)out_size;
  const float* X   = (const float*)d_in[0];
  const float* h0  = (const float*)d_in[1];
  const float* dt  = (const float*)d_in[2];
  const float* W1  = (const float*)d_in[3];
  const float* b1  = (const float*)d_in[4];
  const float* W2  = (const float*)d_in[5];
  const float* b2  = (const float*)d_in[6];
  const float* Wta = (const float*)d_in[7];
  const float* bta = (const float*)d_in[8];
  const float* Wtb = (const float*)d_in[9];
  const float* btb = (const float*)d_in[10];
  float* out = (float*)d_out;

  uint8_t* ws = (uint8_t*)d_ws;
  unsigned*       cnt = (unsigned*)(ws + 0);
  unsigned short* H0  = (unsigned short*)(ws + 1024);
  unsigned short* H1  = (unsigned short*)(ws + 1024 + 131072);
  unsigned*       Wp  = (unsigned*)(ws + 1024 + 262144);       // 16 MiB
  const size_t    base = 1024 + 262144 + 16777216;
  const size_t    xb_bytes = (size_t)BATCH * SEQ * DMODEL * 2; // 128 MiB
  unsigned short* Xb  = (unsigned short*)(ws + base);

  hipLaunchKernelGGL(init_state_kernel, dim3(256), dim3(256), 0, stream, h0, H0, cnt);
  hipLaunchKernelGGL(pack_weights, dim3(16384), dim3(256), 0, stream, W1, W2, Wta, Wtb, Wp);

  if (ws_size >= base + xb_bytes) {
    hipLaunchKernelGGL(convert_x, dim3(32768), dim3(256), 0, stream, X, Xb);
    hipLaunchKernelGGL(cfc_scan_pre, dim3(64), dim3(128), 0, stream,
                       Xb, dt, b1, b2, bta, btb, Wp, H0, H1, out, cnt);
  } else {
    hipLaunchKernelGGL(cfc_scan, dim3(64), dim3(128), 0, stream,
                       X, dt, b1, b2, bta, btb, Wp, H0, H1, out, cnt);
  }
}
